// AttentionTwoLayers_35648228557370
// MI455X (gfx1250) — matrix operand order repaired and checked by reading, not compiled, not run on hardware
//
#include <hip/hip_runtime.h>
#include <hip/hip_bf16.h>
#include <cstdint>
#include <cstddef>

// ---------------------------------------------------------------------------
// AttentionTwoLayers fused pipeline for gfx1250 (MI455X, wave32, WMMA)
//
//   feat_out = (features^T) @ Wv  -> done as  WvT @ features  (M=i, N=hw, K=c)
//   hidden   = tanh(feat_out + (h@Ws + bs) + bv)
//   logits   = hidden . Wl            (bl dropped: softmax shift-invariant)
//   scores   = softmax_hw(logits)
//   weighted = sum_hw features * scores
//
// Round 2: features staging now uses CDNA5 GLOBAL_LOAD_ASYNC_TO_LDS_B128
// (ASYNCcnt-tracked, double-buffered f32 staging) so chunk j+1's HBM traffic
// overlaps chunk j's convert+WMMA phase.
// ---------------------------------------------------------------------------

typedef __attribute__((ext_vector_type(16))) __bf16        v16bf;
typedef __attribute__((ext_vector_type(8)))  float         v8f;
typedef __attribute__((ext_vector_type(4)))  float         v4f;
typedef __attribute__((ext_vector_type(8)))  unsigned short v8us;

#define BATCH 128
#define CH    1024
#define HW    256
#define INTR  512
#define LSTM  512
#define KC    64      // K-chunk of channels staged in LDS
#define KCP   72      // padded k extent (144B row stride, 16B-aligned)
#define NB    64      // hw columns per block
#define NCHUNK (CH / KC)          // 16
#define ASYNC_PER_CHUNK 4         // async b128 instructions per wave per chunk

__device__ __forceinline__ unsigned short f2bf(float f) {
  // round-to-nearest-even f32 -> bf16
  unsigned int u = __float_as_uint(f);
  u += 0x7fffu + ((u >> 16) & 1u);
  return (unsigned short)(u >> 16);
}

__device__ __forceinline__ v16bf frag16(const unsigned short* p0,
                                        const unsigned short* p1) {
  union { v8us u[2]; v16bf b; } cv;
  cv.u[0] = *reinterpret_cast<const v8us*>(p0);   // 16B -> b128 load
  cv.u[1] = *reinterpret_cast<const v8us*>(p1);   // 16B -> b128 load
  return cv.b;
}

// Generic pointer -> 32-bit LDS byte offset (addrspace(3) ptrtoint).
__device__ __forceinline__ unsigned lds_addr(const void* p) {
  return (unsigned)(unsigned long long)
      (__attribute__((address_space(3))) const void*)(p);
}

template <int N>
__device__ __forceinline__ void wait_asynccnt() {
#if defined(__has_builtin) && __has_builtin(__builtin_amdgcn_s_wait_asynccnt)
  __builtin_amdgcn_s_wait_asynccnt(N);
#else
  asm volatile("s_wait_asynccnt %0" ::"n"(N));
#endif
}

// --- Kernel 1: Wv [C,I] f32 -> WvT [I,C] bf16 (A-matrix friendly layout) ----
__global__ __launch_bounds__(256) void prep_wvt(const float* __restrict__ Wv,
                                                unsigned short* __restrict__ WvT) {
  int idx = blockIdx.x * 256 + threadIdx.x;   // 0 .. C*I-1, coalesced read
  int c = idx >> 9;                           // / INTR(512)
  int i = idx & (INTR - 1);
  WvT[(size_t)i * CH + c] = f2bf(Wv[idx]);
}

// --- Kernel 2: state_out[b,i] = h[b,:] @ Ws[:,i] + bs[i] --------------------
__global__ __launch_bounds__(256) void state_gemm(const float* __restrict__ h,
                                                  const float* __restrict__ Ws,
                                                  const float* __restrict__ bs,
                                                  float* __restrict__ out) {
  const int b = blockIdx.y;
  const int i = blockIdx.x * 256 + threadIdx.x;   // 0..511
  const float* hr = h + (size_t)b * LSTM;         // uniform -> scalar loads
  float acc = bs[i];
  for (int k = 0; k < LSTM; ++k)
    acc = fmaf(hr[k], Ws[(size_t)k * INTR + i], acc);  // coalesced in i
  out[(size_t)b * INTR + i] = acc;
}

// --- Kernel 3: fused bf16-WMMA GEMM + bias + tanh + logits dot --------------
// grid = (HW/NB = 4, BATCH), block = 256 threads = 8 waves.
// Wave w owns i-tiles [w*64, w*64+64) x all 4 hw-tiles of this block.
__global__ __launch_bounds__(256) void fused_feat_gemm(
    const float* __restrict__ feats,          // [B, C, HW]
    const unsigned short* __restrict__ WvT,   // bf16 [I, C]
    const float* __restrict__ stateOut,       // [B, I] (includes bs)
    const float* __restrict__ bv,             // [I]
    const float* __restrict__ Wl,             // [I]
    float* __restrict__ logits)               // [B, HW]
{
  __shared__ __align__(16) float sF32[2][KC * NB];       // async f32 staging
  __shared__ __align__(16) unsigned short sB[NB * KCP];  // bf16 [hw][k]
  __shared__ float sState[INTR];
  __shared__ float sWl[INTR];
  __shared__ float sLogits[NB];

  const int b     = blockIdx.y;
  const int hw0   = blockIdx.x * NB;
  const int tid   = threadIdx.x;
  const int wave  = tid >> 5;
  const int lane  = tid & 31;
  const int lrow  = lane & 15;
  const int lhalf = lane >> 4;
  const int i0w   = wave * 64;

  const float* fb = feats + (size_t)b * CH * HW + hw0;

  // Issue one chunk's async copy: 64 rows x 64 f32 (=256B/row), 16B per slot.
  // 1024 slots / 256 threads = 4 async b128 per thread (=4 per wave).
  auto issue_chunk = [&](int k0, int bufsel) {
    const float* gbase = fb + (size_t)k0 * HW;          // wave-uniform
    const unsigned ldsbase = lds_addr(&sF32[bufsel][0]);
#pragma unroll
    for (int it = 0; it < ASYNC_PER_CHUNK; ++it) {
      int idx = it * 256 + tid;                // 16B slot id, 0..1023
      int k = idx >> 4;                        // chunk row
      int cb = (idx & 15) * 16;                // byte offset within row
      unsigned lofs = ldsbase + (unsigned)(k * (NB * 4) + cb);
      unsigned gofs = (unsigned)(k * (HW * 4) + cb);
      asm volatile("global_load_async_to_lds_b128 %0, %1, %2 offset:0"
                   :
                   : "v"(lofs), "v"(gofs), "s"(gbase)
                   : "memory");
    }
  };

  issue_chunk(0, 0);   // prologue: chunk 0 in flight during setup below

  for (int i = tid; i < INTR; i += 256) {
    sState[i] = stateOut[(size_t)b * INTR + i] + bv[i];
    sWl[i]    = Wl[i];
  }
  if (tid < NB) sLogits[tid] = 0.0f;

  v8f acc[4][4];
  {
    v8f z = {};
#pragma unroll
    for (int it = 0; it < 4; ++it)
#pragma unroll
      for (int nt = 0; nt < 4; ++nt) acc[it][nt] = z;
  }

  for (int j = 0; j < NCHUNK; ++j) {
    const int k0 = j * KC;
    if (j + 1 < NCHUNK) {
      issue_chunk(k0 + KC, (j + 1) & 1);   // overlaps convert+compute below
      wait_asynccnt<ASYNC_PER_CHUNK>();    // chunk j arrived; j+1 outstanding
    } else {
      wait_asynccnt<0>();
    }
    __syncthreads();  // all waves' async data visible; all compute(j-1) done

    // Convert chunk j: sF32[j&1] (f32 [k][n]) -> sB (bf16 [n][k], padded).
    {
      const float* src = &sF32[j & 1][0];
#pragma unroll
      for (int it = 0; it < 4; ++it) {
        int base = (it * 256 + tid) * 4;          // 4 consecutive floats
        int k = base >> 6;
        int n0 = base & 63;
        v4f v = *reinterpret_cast<const v4f*>(&src[base]);  // ds b128 load
#pragma unroll
        for (int e = 0; e < 4; ++e)
          sB[(n0 + e) * KCP + k] = f2bf(v[e]);
      }
    }
    if (lane == 0 && j + 1 < NCHUNK)
      __builtin_prefetch(WvT + (size_t)i0w * CH + k0 + KC, 0, 0);
    __syncthreads();

#pragma unroll
    for (int ks = 0; ks < KC; ks += 32) {
      // A fragments (global, bf16 WvT, row-major): per 16-bit A layout,
      // lanes<16 hold K[ks..+7] & K[ks+16..+23]; lanes>=16 K[+8..15],[+24..31]
      v16bf afr[4];
#pragma unroll
      for (int it = 0; it < 4; ++it) {
        const unsigned short* ap =
            WvT + (size_t)(i0w + it * 16 + lrow) * CH + (k0 + ks) + lhalf * 8;
        afr[it] = frag16(ap, ap + 16);
      }
      // B fragments (LDS, transposed chunk): lane holds column hw=n,
      // same K layout as A (lanes<16: K[ks..+7],[+16..23]; lanes>=16: K[+8..15],[+24..31])
      v16bf bfr[4];
#pragma unroll
      for (int nt = 0; nt < 4; ++nt) {
        const unsigned short* bp = &sB[(nt * 16 + lrow) * KCP + ks + lhalf * 8];
        bfr[nt] = frag16(bp, bp + 16);
      }
      // 16 independent WMMAs per K-step -> deep XDL pipelining
#pragma unroll
      for (int it = 0; it < 4; ++it)
#pragma unroll
        for (int nt = 0; nt < 4; ++nt)
          acc[it][nt] = __builtin_amdgcn_wmma_f32_16x16x32_bf16(
              false, afr[it], false, bfr[nt], (short)0, acc[it][nt],
              false, false);
    }
  }

  // Epilogue: hidden = tanh(D + state); partial logits = hidden . Wl over i.
  // C/D layout: VGPR r -> row M = lhalf*8 + r, column N = lrow.
#pragma unroll
  for (int nt = 0; nt < 4; ++nt) {
    float part = 0.0f;
#pragma unroll
    for (int it = 0; it < 4; ++it) {
#pragma unroll
      for (int r = 0; r < 8; ++r) {
        int i = i0w + it * 16 + lhalf * 8 + r;
        float hv = tanhf(acc[it][nt][r] + sState[i]);
        part = fmaf(hv, sWl[i], part);
      }
    }
    part += __shfl_xor(part, 16, 32);          // pair lanes l / l+16 (same hw)
    if (lane < 16) atomicAdd(&sLogits[nt * 16 + lane], part);  // ds_add_f32
  }
  __syncthreads();
  if (tid < NB) logits[(size_t)b * HW + hw0 + tid] = sLogits[tid];
}

// --- Kernel 4: per-batch softmax over HW + weighted spatial pooling ---------
__global__ __launch_bounds__(256) void softmax_pool(
    const float* __restrict__ logits,   // [B, HW]
    const float* __restrict__ feats,    // [B, C, HW]
    float* __restrict__ weighted,       // [B, C]
    float* __restrict__ scores)         // [B, HW]
{
  __shared__ float sS[HW];
  __shared__ float redMax[8];
  __shared__ float redSum[8];
  const int b    = blockIdx.x;
  const int tid  = threadIdx.x;
  const int wave = tid >> 5;
  const int lane = tid & 31;

  float lg = logits[(size_t)b * HW + tid];
  float m = lg;
#pragma unroll
  for (int o = 16; o >= 1; o >>= 1) m = fmaxf(m, __shfl_xor(m, o, 32));
  if (lane == 0) redMax[wave] = m;
  __syncthreads();
  m = redMax[0];
#pragma unroll
  for (int w = 1; w < 8; ++w) m = fmaxf(m, redMax[w]);

  float e = __expf(lg - m);
  float s = e;
#pragma unroll
  for (int o = 16; o >= 1; o >>= 1) s += __shfl_xor(s, o, 32);
  if (lane == 0) redSum[wave] = s;
  __syncthreads();
  s = 0.0f;
#pragma unroll
  for (int w = 0; w < 8; ++w) s += redSum[w];

  float sc = e / s;
  sS[tid] = sc;
  scores[(size_t)b * HW + tid] = sc;
  __syncthreads();

  // weighted[b,c] = sum_hw feats[b,c,hw] * sS[hw]; second features pass (L2).
  const float* fbase = feats + (size_t)b * CH * HW;
  for (int c = wave; c < CH; c += 8) {
    const float* row = fbase + (size_t)c * HW;
    float p = 0.0f;
#pragma unroll
    for (int hh = 0; hh < HW / 32; ++hh)
      p = fmaf(row[hh * 32 + lane], sS[hh * 32 + lane], p);
#pragma unroll
    for (int o = 16; o >= 1; o >>= 1) p += __shfl_xor(p, o, 32);
    if (lane == 0) weighted[(size_t)b * CH + c] = p;
  }
}

// ---------------------------------------------------------------------------
extern "C" void kernel_launch(void* const* d_in, const int* in_sizes, int n_in,
                              void* d_out, int out_size, void* d_ws, size_t ws_size,
                              hipStream_t stream) {
  (void)in_sizes; (void)n_in; (void)out_size; (void)ws_size;

  const float* feats = (const float*)d_in[0];  // [128,1024,16,16]
  const float* hst   = (const float*)d_in[1];  // [128,512]
  const float* Wv    = (const float*)d_in[2];  // [1024,512]
  const float* bv    = (const float*)d_in[3];  // [512]
  const float* Ws    = (const float*)d_in[4];  // [512,512]
  const float* bs    = (const float*)d_in[5];  // [512]
  const float* Wl    = (const float*)d_in[6];  // [512,1]
  // d_in[7] = bl: adding a constant to logits leaves softmax scores unchanged,
  // and bl does not feed the outputs any other way -> intentionally dropped.

  // Workspace layout: WvT bf16 (1 MB) | state_out f32 (256 KB) | logits (128 KB)
  unsigned short* wvt = (unsigned short*)d_ws;
  float* stateOut = (float*)((char*)d_ws + (size_t)INTR * CH * sizeof(unsigned short));
  float* logitsW  = stateOut + (size_t)BATCH * INTR;

  float* weighted = (float*)d_out;                       // [128,1024]
  float* scores   = (float*)d_out + (size_t)BATCH * CH;  // [128,16,16]

  prep_wvt<<<(CH * INTR) / 256, 256, 0, stream>>>(Wv, wvt);
  state_gemm<<<dim3(INTR / 256, BATCH), 256, 0, stream>>>(hst, Ws, bs, stateOut);
  fused_feat_gemm<<<dim3(HW / NB, BATCH), 256, 0, stream>>>(
      feats, wvt, stateOut, bv, Wl, logitsW);
  softmax_pool<<<BATCH, 256, 0, stream>>>(logitsW, feats, weighted, scores);
}
